// CouplingNSF_49821620634130
// MI455X (gfx1250) — matrix operand order repaired and checked by reading, not compiled, not run on hardware
//
#include <hip/hip_runtime.h>
#include <hip/hip_bf16.h>

typedef __attribute__((ext_vector_type(16))) __bf16 v16bf;
typedef __attribute__((ext_vector_type(8)))  float  v8f;

#define BATCH  131072
#define HID    128
#define BTAIL  3.0f
#define MINW   0.001f
#define OBS    388      // padded ob row stride (floats): 16 dims * 24 + 4; ≡4 mod 64 banks

// bf16 weight-fragment offsets inside d_ws (elements)
#define WF_W1  0        // 1 ktile * 8 ntile  * 512 = 4096
#define WF_W2  4096     // 4 * 8  * 512 = 16384
#define WF_W3  20480    // 4 * 8  * 512 = 16384
#define WF_W4  36864    // 4 * 46 * 512 = 94208

// ---------------- weight pre-conversion: f32 row-major -> bf16 B-fragments ---------
// B (32x16 bf16) wave32 layout: lanes 0-15 hold K={0..7,16..23}, lanes 16-31 hold K={8..15,24..31}.
__global__ void wconv_kernel(const float* __restrict__ W, __bf16* __restrict__ dst,
                             int Ntot, int total) {
  int o = blockIdx.x * 256 + threadIdx.x;
  if (o >= total) return;
  int j   = o & 15;
  int L   = (o >> 4) & 31;
  int fi  = o >> 9;              // fragment index = kt*nnt + nt
  int nnt = Ntot >> 4;
  int nt  = fi % nnt;
  int kt  = fi / nnt;
  int n   = nt * 16 + (L & 15);
  int k   = kt * 32 + ((L >> 4) << 3) + (j & 7) + ((j >> 3) << 4);
  dst[o] = (__bf16)W[k * Ntot + n];
}

// ---------------- fragment loaders ----------------
// A (16x32 bf16) wave32 layout (ISA table): lane<16: elems {K0..7, K16..23}; lane>=16: {K8..15, K24..31}
__device__ __forceinline__ v16bf load_A(const __bf16* src, int ldh, int mt, int kt, int lane) {
  const __bf16* base = src + (mt * 16 + (lane & 15)) * ldh + kt * 32 + ((lane >> 4) << 3);
  union { v16bf v; uint4 q[2]; } r;
  r.q[0] = *(const uint4*)(base);        // elems K .. K+7
  r.q[1] = *(const uint4*)(base + 16);   // elems K+16..K+23
  return r.v;
}

__device__ __forceinline__ v16bf load_B(const __bf16* frag0, int fi, int lane) {
  const __bf16* p = frag0 + ((fi << 5) + lane) * 16;  // 32B per lane, coalesced
  union { v16bf v; uint4 q[2]; } r;
  r.q[0] = *(const uint4*)(p);
  r.q[1] = *(const uint4*)(p + 8);
  return r.v;
}

__device__ __forceinline__ v8f bwmma(v16bf a, v16bf b, v8f c) {
  return __builtin_amdgcn_wmma_f32_16x16x32_bf16(false, a, false, b, (short)0, c, false, false);
}

// branchless ELU via hardware v_exp_f32 (arg <= 0 on the taken path)
__device__ __forceinline__ float elu_fast(float v) {
  float ev = __expf(v) - 1.f;
  return v > 0.f ? v : ev;
}

// C/D (16x16 f32): vgpr r, lane L -> D[r + 8*(L>=16)][L&15]
__device__ __forceinline__ void store_elu(__bf16* dst, v8f c, int mt, int nt, int lane,
                                          const float* __restrict__ bias) {
  int col   = nt * 16 + (lane & 15);
  float bn  = bias[col];
  int rbase = mt * 16 + ((lane >> 4) << 3);
#pragma unroll
  for (int r = 0; r < 8; ++r)
    dst[(rbase + r) * HID + col] = (__bf16)elu_fast(c[r] + bn);
}

// ---------------- fused MLP + RQS spline kernel ----------------
__global__ __launch_bounds__(256) void nsf_kernel(
    const float* __restrict__ x,
    const float* __restrict__ b1, const float* __restrict__ b2,
    const float* __restrict__ b3, const float* __restrict__ b4,
    const __bf16* __restrict__ wf,
    float* __restrict__ out) {
  __shared__ __bf16 x1p[64 * 32];
  __shared__ __bf16 h0[64 * HID];
  __shared__ __bf16 h1[64 * HID];
  __shared__ __align__(16) float ob[64 * OBS];   // [row][16 dims][24 padded params]
  __shared__ float  part[256];

  const int tid  = threadIdx.x;
  const int lane = tid & 31;
  const int w    = tid >> 5;
  const int r0   = blockIdx.x * 64;

  // load 64x32 x1 tile (float4) -> bf16 LDS; copy x1 straight to output
#pragma unroll
  for (int i = 0; i < 2; ++i) {
    int idx = tid + i * 256;                     // 512 float4 = 64 rows x 8
    int rl = idx >> 3, c4 = (idx & 7) * 4;
    float4 v = *(const float4*)&x[(r0 + rl) * 64 + c4];
    *(float4*)&out[(r0 + rl) * 64 + c4] = v;
    union { __bf16 h[4]; uint2 u; } cv;
    cv.h[0] = (__bf16)v.x; cv.h[1] = (__bf16)v.y;
    cv.h[2] = (__bf16)v.z; cv.h[3] = (__bf16)v.w;
    *(uint2*)&x1p[rl * 32 + c4] = cv.u;
  }
  __syncthreads();

  // layer 1: [64x32] @ [32x128]   (nt fixed per wave -> B hoisted; 4 independent WMMAs)
#pragma unroll
  for (int t = w; t < 32; t += 8) {
    int mt = t >> 3, nt = t & 7;
    v8f c = {};
    c = bwmma(load_A(x1p, 32, mt, 0, lane), load_B(wf + WF_W1, nt, lane), c);
    store_elu(h0, c, mt, nt, lane, b1);
  }
  __syncthreads();

  // layer 2: [64x128] @ [128x128]
#pragma unroll
  for (int t = w; t < 32; t += 8) {
    int mt = t >> 3, nt = t & 7;
    v8f c = {};
#pragma unroll
    for (int kt = 0; kt < 4; ++kt)
      c = bwmma(load_A(h0, HID, mt, kt, lane), load_B(wf + WF_W2, kt * 8 + nt, lane), c);
    store_elu(h1, c, mt, nt, lane, b2);
  }
  __syncthreads();

  // layer 3
#pragma unroll
  for (int t = w; t < 32; t += 8) {
    int mt = t >> 3, nt = t & 7;
    v8f c = {};
#pragma unroll
    for (int kt = 0; kt < 4; ++kt)
      c = bwmma(load_A(h1, HID, mt, kt, lane), load_B(wf + WF_W3, kt * 8 + nt, lane), c);
    store_elu(h0, c, mt, nt, lane, b3);
  }
  __syncthreads();

  float ldacc = 0.f;
  const int row = tid & 63;
  const int g   = tid >> 6;
  const int grow = r0 + row;

  // layer 4 + spline, two 368-column chunks (16 dims each)
#pragma unroll 1
  for (int ch = 0; ch < 2; ++ch) {
#pragma unroll
    for (int t = w; t < 92; t += 8) {            // 4 mtiles * 23 ntiles
      int mt = t / 23, ntl = t % 23;
      int nt = ch * 23 + ntl;
      v8f c = {};
#pragma unroll
      for (int kt = 0; kt < 4; ++kt)
        c = bwmma(load_A(h0, HID, mt, kt, lane), load_B(wf + WF_W4, kt * 46 + nt, lane), c);
      int col  = ntl * 16 + (lane & 15);         // 0..367 within chunk
      int dl   = col / 23;                       // dim 0..15 (magic-mul division)
      int cofs = dl * 24 + (col - dl * 23);      // padded offset
      float bn  = b4[nt * 16 + (lane & 15)];
      int rbase = mt * 16 + ((lane >> 4) << 3);
#pragma unroll
      for (int r = 0; r < 8; ++r)
        ob[(rbase + r) * OBS + cofs] = c[r] + bn;
    }
    __syncthreads();

    // rational-quadratic spline: this thread handles 4 dims of one row
#pragma unroll 1
    for (int i = 0; i < 4; ++i) {
      int dl = g * 4 + i;                        // local dim in chunk (0..15)
      int dg = ch * 16 + dl;                     // global dim (0..31)
      union { float p[24]; float4 q[6]; } P;     // 6x ds_load_b128 (24th float = pad, unused)
      {
        const float4* src = (const float4*)&ob[row * OBS + dl * 24];
#pragma unroll
        for (int j = 0; j < 6; ++j) P.q[j] = src[j];
      }
      const float* p = P.p;
      float xin = x[grow * 64 + 32 + dg];

      // widths (softmax over p[0..7]) -- args <= 0, native v_exp_f32
      float mx = p[0];
#pragma unroll
      for (int j = 1; j < 8; ++j) mx = fmaxf(mx, p[j]);
      float e[8], s = 0.f;
#pragma unroll
      for (int j = 0; j < 8; ++j) { e[j] = __expf(p[j] - mx); s += e[j]; }
      float inv = 1.f / s;
      float cw[9], wid[8];
      cw[0] = -BTAIL;
      float cum = 0.f;
#pragma unroll
      for (int j = 0; j < 8; ++j) {
        float wj = MINW + (1.f - 8.f * MINW) * (e[j] * inv);
        cum += wj;
        cw[j + 1] = -BTAIL + 2.f * BTAIL * cum;
      }
      cw[8] = BTAIL;
#pragma unroll
      for (int j = 0; j < 8; ++j) wid[j] = cw[j + 1] - cw[j];

      // heights (softmax over p[8..15])
      mx = p[8];
#pragma unroll
      for (int j = 1; j < 8; ++j) mx = fmaxf(mx, p[8 + j]);
      s = 0.f;
#pragma unroll
      for (int j = 0; j < 8; ++j) { e[j] = __expf(p[8 + j] - mx); s += e[j]; }
      inv = 1.f / s;
      float chh[9], hei[8];
      chh[0] = -BTAIL;
      cum = 0.f;
#pragma unroll
      for (int j = 0; j < 8; ++j) {
        float hj = MINW + (1.f - 8.f * MINW) * (e[j] * inv);
        cum += hj;
        chh[j + 1] = -BTAIL + 2.f * BTAIL * cum;
      }
      chh[8] = BTAIL;
#pragma unroll
      for (int j = 0; j < 8; ++j) hei[j] = chh[j + 1] - chh[j];

      // derivatives: softplus via native exp/log; endpoints are exactly 1.0
      float der[9];
      der[0] = 1.f; der[8] = 1.f;
#pragma unroll
      for (int j = 1; j < 8; ++j) {
        float u = p[15 + j];
        float sp = __logf(1.f + __expf(u));
        der[j] = MINW + (u > 20.f ? u : sp);     // cndmask, no branch
      }

      float xc = fminf(fmaxf(xin, -BTAIL), BTAIL);
      int bin = 0;
#pragma unroll
      for (int k = 1; k < 8; ++k) bin += (xc >= cw[k]) ? 1 : 0;

      float icw = cw[0], iw = wid[0], ich = chh[0], ih = hei[0], id0 = der[0], id1 = der[1];
#pragma unroll
      for (int k = 1; k < 8; ++k) {
        bool sel = (bin == k);
        icw = sel ? cw[k] : icw;  iw  = sel ? wid[k] : iw;
        ich = sel ? chh[k] : ich; ih  = sel ? hei[k] : ih;
        id0 = sel ? der[k] : id0; id1 = sel ? der[k + 1] : id1;
      }

      float delta = ih / iw;
      float th  = (xc - icw) / iw;
      float t1  = th * (1.f - th);
      float num = ih * (delta * th * th + id0 * t1);
      float den = delta + (id0 + id1 - 2.f * delta) * t1;
      float yv  = ich + num / den;
      float omt = 1.f - th;
      float dnm = delta * delta * (id1 * th * th + 2.f * delta * t1 + id0 * omt * omt);
      float ldv = __logf(dnm) - 2.f * __logf(den);
      bool inside = (xin >= -BTAIL) && (xin <= BTAIL);
      out[grow * 64 + 32 + dg] = inside ? yv : xin;
      ldacc += inside ? ldv : 0.f;
    }
    __syncthreads();                             // before chunk 1 overwrites ob
  }

  part[tid] = ldacc;
  __syncthreads();
  if (tid < 64)
    out[BATCH * 64 + r0 + tid] =
        part[tid] + part[tid + 64] + part[tid + 128] + part[tid + 192];
}

// ---------------- launch ----------------
extern "C" void kernel_launch(void* const* d_in, const int* in_sizes, int n_in,
                              void* d_out, int out_size, void* d_ws, size_t ws_size,
                              hipStream_t stream) {
  const float* x  = (const float*)d_in[0];
  const float* W1 = (const float*)d_in[1];
  const float* B1 = (const float*)d_in[2];
  const float* W2 = (const float*)d_in[3];
  const float* B2 = (const float*)d_in[4];
  const float* W3 = (const float*)d_in[5];
  const float* B3 = (const float*)d_in[6];
  const float* W4 = (const float*)d_in[7];
  const float* B4 = (const float*)d_in[8];
  __bf16* wf = (__bf16*)d_ws;

  wconv_kernel<<<(4096 + 255) / 256, 256, 0, stream>>>(W1, wf + WF_W1, 128, 4096);
  wconv_kernel<<<(16384 + 255) / 256, 256, 0, stream>>>(W2, wf + WF_W2, 128, 16384);
  wconv_kernel<<<(16384 + 255) / 256, 256, 0, stream>>>(W3, wf + WF_W3, 128, 16384);
  wconv_kernel<<<(94208 + 255) / 256, 256, 0, stream>>>(W4, wf + WF_W4, 736, 94208);

  nsf_kernel<<<BATCH / 64, 256, 0, stream>>>(x, B1, B2, B3, B4, wf, (float*)d_out);
}
